// ContextAwareAttention_24902220382495
// MI455X (gfx1250) — matrix operand order repaired and checked by reading, not compiled, not run on hardware
//
#include <hip/hip_runtime.h>
#include <cstdint>
#include <cstddef>

#define DIMQ   256
#define DIM2   512
#define BATCH  4
#define SEQ    4096
#define MTOT   (BATCH*SEQ)     // 16384
#define QSCALE 0.0625f         // DIM^-0.5

typedef __bf16 v16bf __attribute__((ext_vector_type(16)));
typedef __bf16 v8bf  __attribute__((ext_vector_type(8)));
typedef float  v8f   __attribute__((ext_vector_type(8)));
typedef unsigned int v4u __attribute__((ext_vector_type(4)));
typedef unsigned int v8u __attribute__((ext_vector_type(8)));

union V16U { v16bf v; v8bf h[2]; };

__device__ __forceinline__ v8f wmma_bf16(v16bf a, v16bf b, v8f c) {
  return __builtin_amdgcn_wmma_f32_16x16x32_bf16(false, a, false, b, (short)0, c,
                                                 false, false);
}

// A fragment (16x32 bf16, MxK) from row-major matrix, tile origin (m0,k0).
// lanes 0-15: rows m0..m0+15, K 0..15; lanes 16-31: same rows, K 16..31
__device__ __forceinline__ v16bf load_a_frag(const __bf16* __restrict__ A, int lda,
                                             int m0, int k0) {
  int lane = threadIdx.x & 31;
  int m = m0 + (lane & 15);
  int kh = (lane >> 4) * 16;
  const __bf16* p = A + (size_t)m * lda + k0 + kh;
  V16U u;
  u.h[0] = *(const v8bf*)p;
  u.h[1] = *(const v8bf*)(p + 8);
  return u.v;
}

// B fragment (32x16 bf16, KxN), B[k][n] = W[n0+n][k0+k] from row-major W (x@W.T form).
// lanes 0-15: col n, K 0..15 ; lanes 16-31: col n, K 16..31 -> one contiguous 32B read
__device__ __forceinline__ v16bf load_b_frag(const __bf16* __restrict__ W, int ldw,
                                             int n0, int k0) {
  int lane = threadIdx.x & 31;
  int n = n0 + (lane & 15);
  int kb = k0 + (lane >> 4) * 16;
  return *(const v16bf*)(W + (size_t)n * ldw + kb);
}

// ---------------------------------------------------------------- conversion
__global__ void cvt_bf16_kernel(const float* __restrict__ src,
                                __bf16* __restrict__ dst, int n, float scale) {
  int i = blockIdx.x * blockDim.x + threadIdx.x;
  int stride = gridDim.x * blockDim.x;
  for (; i < n; i += stride) dst[i] = (__bf16)(src[i] * scale);
}

// ------------------------------------------------------- gate = sigmoid(ctx@wc.T+bc)
__global__ __launch_bounds__(256)
void gate_gemm_kernel(const __bf16* __restrict__ ctx, const __bf16* __restrict__ wc,
                      const float* __restrict__ bc, float* __restrict__ gate) {
  int wave = threadIdx.x >> 5;
  int lane = threadIdx.x & 31;
  int m0 = blockIdx.x * 128 + wave * 16;
  for (int nt = 0; nt < DIM2 / 16; ++nt) {
    int n0 = nt * 16;
    v8f acc = {};
#pragma unroll
    for (int k0 = 0; k0 < DIM2; k0 += 32)
      acc = wmma_bf16(load_a_frag(ctx, DIM2, m0, k0),
                      load_b_frag(wc, DIM2, n0, k0), acc);
    int n = n0 + (lane & 15);
    float bias = bc[n];
#pragma unroll
    for (int r = 0; r < 8; ++r) {
      int m = m0 + r + 8 * (lane >> 4);
      float x = acc[r] + bias;
      gate[(size_t)m * DIM2 + n] = 1.0f / (1.0f + __expf(-x));
    }
  }
}

// ---------------- out = A1@W1.T + A2@W2.T  (complex-linear halves, W2 pre-negated
// by the host when a subtraction is needed).  mode 0: q (gate*SCALE, row-major at
// column coff); mode 1: k (row-major at coff); mode 2: v (store transposed [n][m]).
__global__ __launch_bounds__(256)
void proj_gemm_kernel(const __bf16* __restrict__ A1, const __bf16* __restrict__ W1,
                      const __bf16* __restrict__ A2, const __bf16* __restrict__ W2,
                      const float* __restrict__ gate, __bf16* __restrict__ out,
                      int mode, int coff) {
  int wave = threadIdx.x >> 5;
  int lane = threadIdx.x & 31;
  int m0 = blockIdx.x * 128 + wave * 16;
  for (int nt = 0; nt < DIMQ / 16; ++nt) {
    int n0 = nt * 16;
    v8f acc = {};
#pragma unroll
    for (int k0 = 0; k0 < DIMQ; k0 += 32)
      acc = wmma_bf16(load_a_frag(A1, DIMQ, m0, k0),
                      load_b_frag(W1, DIMQ, n0, k0), acc);
#pragma unroll
    for (int k0 = 0; k0 < DIMQ; k0 += 32)
      acc = wmma_bf16(load_a_frag(A2, DIMQ, m0, k0),
                      load_b_frag(W2, DIMQ, n0, k0), acc);
    int n = n0 + (lane & 15);
#pragma unroll
    for (int r = 0; r < 8; ++r) {
      int m = m0 + r + 8 * (lane >> 4);
      float v = acc[r];
      if (mode == 0) {
        v *= gate[(size_t)m * DIM2 + coff + n] * QSCALE;
        out[(size_t)m * DIM2 + coff + n] = (__bf16)v;
      } else if (mode == 1) {
        out[(size_t)m * DIM2 + coff + n] = (__bf16)v;
      } else {
        out[(size_t)n * MTOT + m] = (__bf16)v;   // v stored transposed [d][token]
      }
    }
  }
}

// ------------------------------------------------------------ flash attention
// block = 4 waves; wave w -> (query tile qt = w>>1 of 2, half = w&1: 0=real,1=imag).
// Each wave: 16 queries x 256 output cols, O = 16 v8f accumulators.
__global__ __launch_bounds__(128)
void attn_kernel(const __bf16* __restrict__ qf, const __bf16* __restrict__ kf,
                 const __bf16* __restrict__ vrt, const __bf16* __restrict__ vit,
                 float* __restrict__ out) {
  __shared__ __bf16 Qs[32 * DIM2];     // 32 KB query block
  __shared__ __bf16 Ps[4][16 * 32];    // per-wave P tile (C-layout -> A-layout bounce)

  int b = blockIdx.y;
  int s0 = blockIdx.x * 32;
  int wave = threadIdx.x >> 5;
  int lane = threadIdx.x & 31;
  int qt = wave >> 1;
  int half = wave & 1;

  const __bf16* qsrc = qf + (size_t)(b * SEQ + s0) * DIM2;

  // ---- rows 0..15 of the Q block: Tensor Data Mover (2D tile 16 x 512 bf16)
  if (wave == 0) {
    unsigned lds_off = (unsigned)(uintptr_t)(&Qs[0]);
    unsigned long long ga = (unsigned long long)(uintptr_t)qsrc;
    v4u g0;
    g0[0] = 1u;                                            // count=1, user D#
    g0[1] = lds_off;                                       // lds_addr
    g0[2] = (unsigned)ga;                                  // global_addr[31:0]
    g0[3] = (unsigned)((ga >> 32) & 0x01FFFFFFull) | 0x80000000u; // addr[56:32]|type=2
    v8u g1;
    g1[0] = 1u << 16;     // workgroup_mask=0, data_size=1 (2 bytes), no flags
    g1[1] = 512u << 16;   // tensor_dim0[15:0]=512 (bits 63:48)
    g1[2] = 16u << 16;    // tensor_dim0[31:16]=0 | tensor_dim1[15:0]=16
    g1[3] = 512u << 16;   // tensor_dim1[31:16]=0 | tile_dim0=512
    g1[4] = 16u;          // tile_dim1=16, tile_dim2=0
    g1[5] = 512u;         // tensor_dim0_stride[31:0]=512
    g1[6] = 0u;           // stride0[47:32]=0, stride1[15:0]=0
    g1[7] = 0u;           // stride1[47:16]=0
    asm volatile("tensor_load_to_lds %0, %1" :: "s"(g0), "s"(g1) : "memory");
  }

  // ---- rows 16..31 of the Q block: async global->LDS copies (contiguous 16 KB)
  for (int i = threadIdx.x; i < 16 * DIM2 / 8; i += blockDim.x) {
    unsigned lds_off = (unsigned)(uintptr_t)(&Qs[16 * DIM2 + i * 8]);
    unsigned long long ga =
        (unsigned long long)(uintptr_t)(qsrc + 16 * DIM2 + i * 8);
    asm volatile("global_load_async_to_lds_b128 %0, %1, off"
                 :: "v"(lds_off), "v"(ga) : "memory");
  }
  asm volatile("s_wait_asynccnt 0" ::: "memory");
  if (wave == 0) __builtin_amdgcn_s_wait_tensorcnt(0);
  __syncthreads();

  v8f O[16];
  v8f zero = {};
#pragma unroll
  for (int j = 0; j < 16; ++j) O[j] = zero;
  float mrow[8], lrow[8];
#pragma unroll
  for (int r = 0; r < 8; ++r) { mrow[r] = -1e30f; lrow[r] = 0.0f; }

  const __bf16* vt = half ? vit : vrt;
  const __bf16* krow = kf + (size_t)b * SEQ * DIM2;
  __bf16* ps = &Ps[wave][0];

  for (int t0 = 0; t0 < SEQ; t0 += 32) {
    // ---- prefetch next K tile (one row per lane) to overlap with the WMMA chain
    if (t0 + 32 < SEQ)
      __builtin_prefetch(krow + (size_t)(t0 + 32 + lane) * DIM2, 0, 1);

    // ---- scores: two 16x16 tiles over full 512-dim contraction
    v8f S0 = zero, S1 = zero;
#pragma unroll
    for (int d0 = 0; d0 < DIM2; d0 += 32) {
      int m = qt * 16 + (lane & 15);
      int kh = (lane >> 4) * 16;
      const __bf16* qp = &Qs[m * DIM2 + d0 + kh];
      V16U qa;
      qa.h[0] = *(const v8bf*)qp;
      qa.h[1] = *(const v8bf*)(qp + 8);
      S0 = wmma_bf16(qa.v, load_b_frag(krow, DIM2, t0, d0), S0);
      S1 = wmma_bf16(qa.v, load_b_frag(krow, DIM2, t0 + 16, d0), S1);
    }
    // ---- online softmax (rows live on 16-lane halves: row = r + 8*(lane>>4))
    float alpha[8];
#pragma unroll
    for (int r = 0; r < 8; ++r) {
      float t = fmaxf(S0[r], S1[r]);
#pragma unroll
      for (int msk = 8; msk >= 1; msk >>= 1)
        t = fmaxf(t, __shfl_xor(t, msk, 32));
      float mnew = fmaxf(mrow[r], t);
      alpha[r] = __expf(mrow[r] - mnew);
      mrow[r] = mnew;
      float p0 = __expf(S0[r] - mnew);
      float p1 = __expf(S1[r] - mnew);
      S0[r] = p0; S1[r] = p1;
      float rs = p0 + p1;
#pragma unroll
      for (int msk = 8; msk >= 1; msk >>= 1)
        rs += __shfl_xor(rs, msk, 32);
      lrow[r] = lrow[r] * alpha[r] + rs;
    }
#pragma unroll
    for (int j = 0; j < 16; ++j)
#pragma unroll
      for (int r = 0; r < 8; ++r) O[j][r] *= alpha[r];

    // ---- P: C-layout -> row-major 16x32 in wave-private LDS
#pragma unroll
    for (int r = 0; r < 8; ++r) {
      int m = r + 8 * (lane >> 4);
      int t = lane & 15;
      ps[m * 32 + t]      = (__bf16)S0[r];
      ps[m * 32 + 16 + t] = (__bf16)S1[r];
    }
    // ---- read back as 16x32 A fragment and do P x V
    {
      int m = lane & 15;
      int kh = (lane >> 4) * 16;
      V16U pa;
      pa.h[0] = *(const v8bf*)&ps[m * 32 + kh];
      pa.h[1] = *(const v8bf*)&ps[m * 32 + 8 + kh];
#pragma unroll
      for (int j = 0; j < 16; ++j) {
        int d = j * 16 + (lane & 15);
        int tt = t0 + 16 * (lane >> 4);
        v16bf bv = *(const v16bf*)(vt + (size_t)d * MTOT + b * SEQ + tt);
        O[j] = wmma_bf16(pa.v, bv, O[j]);
      }
    }
  }
  // ---- epilogue: O / l, fp32 out, [.. ,0:256]=real, [..,256:512]=imag
#pragma unroll
  for (int r = 0; r < 8; ++r) {
    float inv = 1.0f / lrow[r];
    int m = s0 + qt * 16 + r + 8 * (lane >> 4);
#pragma unroll
    for (int j = 0; j < 16; ++j) {
      int n = half * DIMQ + j * 16 + (lane & 15);
      out[((size_t)(b * SEQ) + m) * DIM2 + n] = O[j][r] * inv;
    }
  }
}

// ----------------------------------------------------------------- launcher
extern "C" void kernel_launch(void* const* d_in, const int* in_sizes, int n_in,
                              void* d_out, int out_size, void* d_ws, size_t ws_size,
                              hipStream_t stream) {
  (void)in_sizes; (void)n_in; (void)out_size; (void)ws_size;
  const float* z_real  = (const float*)d_in[0];
  const float* z_imag  = (const float*)d_in[1];
  const float* context = (const float*)d_in[2];
  const float* wq_r = (const float*)d_in[3];
  const float* wq_i = (const float*)d_in[4];
  const float* wk_r = (const float*)d_in[5];
  const float* wk_i = (const float*)d_in[6];
  const float* wv_r = (const float*)d_in[7];
  const float* wv_i = (const float*)d_in[8];
  const float* wc   = (const float*)d_in[9];
  const float* bc   = (const float*)d_in[10];
  float* out = (float*)d_out;

  char* ws = (char*)d_ws;
  size_t off = 0;
  auto alloc = [&](size_t bytes) {
    void* p = ws + off;
    off += (bytes + 255) & ~(size_t)255;
    return p;
  };
  __bf16* zr_bf  = (__bf16*)alloc((size_t)MTOT * DIMQ * 2);
  __bf16* zi_bf  = (__bf16*)alloc((size_t)MTOT * DIMQ * 2);
  __bf16* ctx_bf = (__bf16*)alloc((size_t)MTOT * DIM2 * 2);
  __bf16* qflat  = (__bf16*)alloc((size_t)MTOT * DIM2 * 2);
  __bf16* kflat  = (__bf16*)alloc((size_t)MTOT * DIM2 * 2);
  __bf16* vrt    = (__bf16*)alloc((size_t)DIMQ * MTOT * 2);
  __bf16* vit    = (__bf16*)alloc((size_t)DIMQ * MTOT * 2);
  float*  gate   = (float*)alloc((size_t)MTOT * DIM2 * 4);
  __bf16* wqr  = (__bf16*)alloc(DIMQ * DIMQ * 2);
  __bf16* wqi  = (__bf16*)alloc(DIMQ * DIMQ * 2);
  __bf16* wqin = (__bf16*)alloc(DIMQ * DIMQ * 2);
  __bf16* wkr  = (__bf16*)alloc(DIMQ * DIMQ * 2);
  __bf16* wki  = (__bf16*)alloc(DIMQ * DIMQ * 2);
  __bf16* wkin = (__bf16*)alloc(DIMQ * DIMQ * 2);
  __bf16* wvr  = (__bf16*)alloc(DIMQ * DIMQ * 2);
  __bf16* wvi  = (__bf16*)alloc(DIMQ * DIMQ * 2);
  __bf16* wvin = (__bf16*)alloc(DIMQ * DIMQ * 2);
  __bf16* wc_bf = (__bf16*)alloc(DIM2 * DIM2 * 2);

  auto cvt = [&](const float* s, __bf16* d, int n, float sc) {
    int grid = (n + 255) / 256;
    hipLaunchKernelGGL(cvt_bf16_kernel, dim3(grid), dim3(256), 0, stream, s, d, n, sc);
  };
  cvt(z_real, zr_bf, MTOT * DIMQ, 1.0f);
  cvt(z_imag, zi_bf, MTOT * DIMQ, 1.0f);
  cvt(context, ctx_bf, MTOT * DIM2, 1.0f);
  cvt(wq_r, wqr, DIMQ * DIMQ, 1.0f);
  cvt(wq_i, wqi, DIMQ * DIMQ, 1.0f);
  cvt(wq_i, wqin, DIMQ * DIMQ, -1.0f);
  cvt(wk_r, wkr, DIMQ * DIMQ, 1.0f);
  cvt(wk_i, wki, DIMQ * DIMQ, 1.0f);
  cvt(wk_i, wkin, DIMQ * DIMQ, -1.0f);
  cvt(wv_r, wvr, DIMQ * DIMQ, 1.0f);
  cvt(wv_i, wvi, DIMQ * DIMQ, 1.0f);
  cvt(wv_i, wvin, DIMQ * DIMQ, -1.0f);
  cvt(wc, wc_bf, DIM2 * DIM2, 1.0f);

  hipLaunchKernelGGL(gate_gemm_kernel, dim3(MTOT / 128), dim3(256), 0, stream,
                     ctx_bf, wc_bf, bc, gate);

  // q_r = zr@wqr.T - zi@wqi.T ; q_i = zi@wqr.T + zr@wqi.T   (gated, SCALE folded)
  hipLaunchKernelGGL(proj_gemm_kernel, dim3(MTOT / 128), dim3(256), 0, stream,
                     zr_bf, wqr, zi_bf, wqin, gate, qflat, 0, 0);
  hipLaunchKernelGGL(proj_gemm_kernel, dim3(MTOT / 128), dim3(256), 0, stream,
                     zi_bf, wqr, zr_bf, wqi, gate, qflat, 0, DIMQ);
  // k
  hipLaunchKernelGGL(proj_gemm_kernel, dim3(MTOT / 128), dim3(256), 0, stream,
                     zr_bf, wkr, zi_bf, wkin, gate, kflat, 1, 0);
  hipLaunchKernelGGL(proj_gemm_kernel, dim3(MTOT / 128), dim3(256), 0, stream,
                     zi_bf, wkr, zr_bf, wki, gate, kflat, 1, DIMQ);
  // v (stored transposed [d][token])
  hipLaunchKernelGGL(proj_gemm_kernel, dim3(MTOT / 128), dim3(256), 0, stream,
                     zr_bf, wvr, zi_bf, wvin, gate, vrt, 2, 0);
  hipLaunchKernelGGL(proj_gemm_kernel, dim3(MTOT / 128), dim3(256), 0, stream,
                     zi_bf, wvr, zr_bf, wvi, gate, vit, 2, 0);

  hipLaunchKernelGGL(attn_kernel, dim3(SEQ / 32, BATCH), dim3(128), 0, stream,
                     qflat, kflat, vrt, vit, out);
}
